// AttentionFlow_53455162966587
// MI455X (gfx1250) — matrix operand order repaired and checked by reading, not compiled, not run on hardware
//
#include <hip/hip_runtime.h>
#include <hip/hip_bf16.h>

// ---------------------------------------------------------------------------
// Problem constants (match reference)
// ---------------------------------------------------------------------------
#define BB 32
#define NN 50000
#define EE 200000
#define DD 64
#define DLG 256
#define MALL 100000
#define MMEM 80000
#define NSEG 80000

typedef __attribute__((ext_vector_type(16))) _Float16 v16h;
typedef __attribute__((ext_vector_type(8)))  _Float16 v8h;
typedef __attribute__((ext_vector_type(4)))  _Float16 v4h;
typedef __attribute__((ext_vector_type(8)))  float    v8f;

// ---------------------------------------------------------------------------
// Branch-free fast tanh: tanh(x) = sign(x) * (1 - e^{-2|x|}) / (1 + e^{-2|x|})
// Uses native v_exp_f32 / v_rcp_f32; no exec-mask branches, no spills.
// ---------------------------------------------------------------------------
__device__ __forceinline__ float fast_tanh(float x) {
    const float a = __builtin_fabsf(x);
    const float t = __expf(-2.0f * a);
    const float r = (1.0f - t) * __builtin_amdgcn_rcpf(1.0f + t);
    return __builtin_copysignf(r, x);
}

// ---------------------------------------------------------------------------
// WMMA fragment loaders (wave32, V_WMMA_F32_16X16X32_F16 layouts per ISA 7.12.2)
//
// A 16x32 f16 (row-major LDS, stride in halves): lane l -> row M=l&15,
//   halves 0-7  = K [8g   .. 8g+7 ]   (g = l>>4)   -> one 16B contiguous run
//   halves 8-15 = K [16+8g.. 16+8g+7]              -> one 16B contiguous run
// => two ds_load_b128 per fragment.
//
// B 32x16 f16 from TRANSPOSED weights Wt[n][k] (ldk = K length):
//   lane l -> col N=l&15, halves 0-7 = K [k0+8g..+7], halves 8-15 = K [k0+16+8g..+7]
// => two ds_load_b128 per fragment.
//
// C/D 16x16 f32: vgpr r, lane l -> M = r + 8*(l>>4), N = l&15
// ---------------------------------------------------------------------------
__device__ __forceinline__ v16h load_A_frag(const _Float16* __restrict__ base,
                                            int stride, int k0, int lane) {
    const int g = (lane >> 4) & 1;
    const int m = lane & 15;
    const _Float16* p = base + m * stride + k0 + 8 * g;
    v8h lo = *(const v8h*)(p);
    v8h hi = *(const v8h*)(p + 16);
    return __builtin_shufflevector(lo, hi, 0, 1, 2, 3, 4, 5, 6, 7,
                                   8, 9, 10, 11, 12, 13, 14, 15);
}

__device__ __forceinline__ v16h load_B_fragT(const _Float16* __restrict__ Wt,
                                             int ldk, int k0, int n0, int lane) {
    const int g = (lane >> 4) & 1;
    const int n = lane & 15;
    const _Float16* p = Wt + (n0 + n) * ldk + k0 + 8 * g;
    v8h lo = *(const v8h*)(p);
    v8h hi = *(const v8h*)(p + 16);
    return __builtin_shufflevector(lo, hi, 0, 1, 2, 3, 4, 5, 6, 7,
                                   8, 9, 10, 11, 12, 13, 14, 15);
}

#define WMMA_F16(a, b, c) \
    __builtin_amdgcn_wmma_f32_16x16x32_f16(false, (a), false, (b), (short)0, (c), false, false)

// ---------------------------------------------------------------------------
// Zero-fill kernel (d_out + workspace accumulators must be zeroed every call)
// ---------------------------------------------------------------------------
__global__ void zero_kernel(float* __restrict__ p, int n) {
    int i = blockIdx.x * blockDim.x + threadIdx.x;
    if (i < n) p[i] = 0.0f;
}

// ---------------------------------------------------------------------------
// qv = tanh(qh @ W_hlg + b_hlg) + tanh(qr @ W_rlg + b_rlg)   (32 x 64, tiny)
// ---------------------------------------------------------------------------
__global__ void qv_kernel(const float* __restrict__ qh, const float* __restrict__ qr,
                          const float* __restrict__ Whlg, const float* __restrict__ bhlg,
                          const float* __restrict__ Wrlg, const float* __restrict__ brlg,
                          float* __restrict__ qv) {
    const int b = blockIdx.x;
    const int n = threadIdx.x;
    float s1 = bhlg[n], s2 = brlg[n];
    const float* qhb = qh + b * DLG;
    const float* qrb = qr + b * DLG;
#pragma unroll 4
    for (int k = 0; k < DLG; ++k) {
        s1 += qhb[k] * Whlg[k * DD + n];
        s2 += qrb[k] * Wrlg[k * DD + n];
    }
    qv[b * DD + n] = fast_tanh(s1) + fast_tanh(s2);
}

// ---------------------------------------------------------------------------
// Generic WMMA GEMM: Out[out_idx ? out_idx[r] : r] = tanh(X[r] @ W + bias)
// W is transposed into LDS (Wt[n][k]) so B fragments are b128 loads.
// A fragments hoisted across the 4 N-tiles; N-tile loop kept rolled to cap
// live accumulators (no spills).
// ---------------------------------------------------------------------------
template <int DIN, int WAVES>
__global__ __launch_bounds__(WAVES * 32)
void gemm_tanh_kernel(const float* __restrict__ X, const float* __restrict__ W,
                      const float* __restrict__ bias,
                      const int* __restrict__ out_idx,
                      float* __restrict__ Out, int rows) {
    __shared__ _Float16 sWT[DD * DIN];   // transposed: [n][k]
    __shared__ float    sB[DD];
    __shared__ _Float16 sX[WAVES * 16 * DIN];

    const int tid  = threadIdx.x;
    const int lane = tid & 31;
    const int wave = tid >> 5;
    const int nthr = WAVES * 32;

    for (int i = tid; i < DIN * DD; i += nthr) {
        const int k = i >> 6, n = i & 63;          // W row-major (k, n)
        sWT[n * DIN + k] = (_Float16)W[i];
    }
    for (int i = tid; i < DD; i += nthr) sB[i] = bias[i];

    const int row_base = blockIdx.x * (WAVES * 16) + wave * 16;
    _Float16* myX = sX + wave * 16 * DIN;
    for (int i = lane; i < 16 * (DIN / 4); i += 32) {
        const int m = i / (DIN / 4), kq = i - m * (DIN / 4);
        int r = row_base + m;
        if (r >= rows) r = rows - 1;               // clamp (stores are guarded)
        float4 x = *(const float4*)(X + (long)r * DIN + kq * 4);
        v4h h; h[0] = (_Float16)x.x; h[1] = (_Float16)x.y;
               h[2] = (_Float16)x.z; h[3] = (_Float16)x.w;
        *(v4h*)(myX + m * DIN + kq * 4) = h;
    }
    __syncthreads();

    v16h afrag[DIN / 32];
#pragma unroll
    for (int ks = 0; ks < DIN / 32; ++ks)
        afrag[ks] = load_A_frag(myX, DIN, ks * 32, lane);

    const int g    = (lane >> 4) & 1;
    const int nloc = lane & 15;
#pragma unroll 1
    for (int t = 0; t < 4; ++t) {
        v8f c = {};
#pragma unroll
        for (int ks = 0; ks < DIN / 32; ++ks) {
            v16h b = load_B_fragT(sWT, DIN, ks * 32, t * 16, lane);
            c = WMMA_F16(afrag[ks], b, c);
        }
        const int n  = t * 16 + nloc;
        const float bn = sB[n];
#pragma unroll
        for (int r = 0; r < 8; ++r) {
            const int row = row_base + r + 8 * g;
            if (row < rows) {
                const int orow = out_idx ? out_idx[row] : row;
                Out[(long)orow * DD + n] = fast_tanh(c[r] + bn);
            }
        }
    }
}

// ---------------------------------------------------------------------------
// Fused per-edge bilinear scoring:
//   h_vi = hc_full[e2vi] + hu[vi]; h_vj = hc_full[e2vj] + hu[vj]; q = qv[eg]
//   logits = rowsum(tanh(hvi@Wl0+b)*tanh(hvj@Wr0+b))
//          + rowsum(tanh([hvi|q]@Wl1+b)*tanh([hvj|q]@Wr1+b))
// Each wave scores 16 edges (M=16 WMMA tiles); weights transposed in LDS.
// N-tile loop kept rolled: only 4 v8f accumulators live -> no spills.
// ---------------------------------------------------------------------------
#define EWAVES 2

__global__ __launch_bounds__(EWAVES * 32)
void edge_logits_kernel(const int* __restrict__ edges,
                        const float* __restrict__ hc_full,
                        const float* __restrict__ hu,
                        const float* __restrict__ qv,
                        const float* __restrict__ Wl0, const float* __restrict__ bl0,
                        const float* __restrict__ Wr0, const float* __restrict__ br0,
                        const float* __restrict__ Wl1, const float* __restrict__ bl1,
                        const float* __restrict__ Wr1, const float* __restrict__ br1,
                        float* __restrict__ logits, int nE) {
    extern __shared__ unsigned char smem_raw[];
    _Float16* sWl0T = (_Float16*)smem_raw;           // [n:64][k:64]
    _Float16* sWr0T = sWl0T + DD * DD;               // [n:64][k:64]
    _Float16* sWl1T = sWr0T + DD * DD;               // [n:64][k:128]
    _Float16* sWr1T = sWl1T + 2 * DD * DD;           // [n:64][k:128]
    float*    sBia  = (float*)(sWr1T + 2 * DD * DD); // 4*64
    _Float16* sStg  = (_Float16*)(sBia + 4 * DD);    // EWAVES * 3 * 16 * 64

    const int tid  = threadIdx.x;
    const int lane = tid & 31;
    const int wave = tid >> 5;
    const int nthr = EWAVES * 32;

    for (int i = tid; i < DD * DD; i += nthr) {      // (k,n) -> [n][k]
        const int k = i >> 6, n = i & 63;
        sWl0T[n * DD + k] = (_Float16)Wl0[i];
        sWr0T[n * DD + k] = (_Float16)Wr0[i];
    }
    for (int i = tid; i < 2 * DD * DD; i += nthr) {  // k in [0,128)
        const int k = i >> 6, n = i & 63;
        sWl1T[n * 2 * DD + k] = (_Float16)Wl1[i];
        sWr1T[n * 2 * DD + k] = (_Float16)Wr1[i];
    }
    for (int i = tid; i < DD; i += nthr) {
        sBia[i] = bl0[i]; sBia[DD + i] = br0[i];
        sBia[2 * DD + i] = bl1[i]; sBia[3 * DD + i] = br1[i];
    }

    const int ebase = blockIdx.x * (EWAVES * 16) + wave * 16;
    _Float16* Hvi = sStg + wave * 3 * 16 * DD;
    _Float16* Hvj = Hvi + 16 * DD;
    _Float16* Qe  = Hvj + 16 * DD;

    // Stage the wave's 16 gathered rows as f16. Lane l: edge m=l&15, K-half g.
    const int g = (lane >> 4) & 1;
    const int m = lane & 15;
    {
        int e = ebase + m;
        if (e >= nE) e = nE - 1;
        const int* er = edges + (long)e * 8;
        const int eg = er[0], vi = er[1], vj = er[2], e2vi = er[6], e2vj = er[7];
        const float* hic = hc_full + (long)e2vi * DD;
        const float* hiu = hu + (long)vi * DD;
        const float* hjc = hc_full + (long)e2vj * DD;
        const float* hju = hu + (long)vj * DD;
        const float* qrw = qv + eg * DD;
#pragma unroll
        for (int j = 0; j < 8; ++j) {
            const int k = g * 32 + j * 4;
            float4 a  = *(const float4*)(hic + k);
            float4 b  = *(const float4*)(hiu + k);
            float4 c  = *(const float4*)(hjc + k);
            float4 d  = *(const float4*)(hju + k);
            float4 qq = *(const float4*)(qrw + k);
            v4h hi4; hi4[0] = (_Float16)(a.x + b.x); hi4[1] = (_Float16)(a.y + b.y);
                     hi4[2] = (_Float16)(a.z + b.z); hi4[3] = (_Float16)(a.w + b.w);
            v4h hj4; hj4[0] = (_Float16)(c.x + d.x); hj4[1] = (_Float16)(c.y + d.y);
                     hj4[2] = (_Float16)(c.z + d.z); hj4[3] = (_Float16)(c.w + d.w);
            v4h qe4; qe4[0] = (_Float16)qq.x; qe4[1] = (_Float16)qq.y;
                     qe4[2] = (_Float16)qq.z; qe4[3] = (_Float16)qq.w;
            *(v4h*)(Hvi + m * DD + k) = hi4;
            *(v4h*)(Hvj + m * DD + k) = hj4;
            *(v4h*)(Qe  + m * DD + k) = qe4;
        }
    }
    __syncthreads();

    // Hoisted A fragments (reused across all 4 N-tiles and both GEMM pairs;
    // the [h|q] concat's K=64..127 region is exactly Qe's fragments).
    v16h aHvi0 = load_A_frag(Hvi, DD, 0, lane), aHvi1 = load_A_frag(Hvi, DD, 32, lane);
    v16h aHvj0 = load_A_frag(Hvj, DD, 0, lane), aHvj1 = load_A_frag(Hvj, DD, 32, lane);
    v16h aQe0  = load_A_frag(Qe,  DD, 0, lane), aQe1  = load_A_frag(Qe,  DD, 32, lane);

    float partial[8];
#pragma unroll
    for (int r = 0; r < 8; ++r) partial[r] = 0.0f;

    const int nloc = lane & 15;
#pragma unroll 1
    for (int t = 0; t < 4; ++t) {
        const int n0 = t * 16;
        v8f l0c = {}, r0c = {}, l1c = {}, r1c = {};
        l0c = WMMA_F16(aHvi0, load_B_fragT(sWl0T, DD, 0,  n0, lane), l0c);
        l0c = WMMA_F16(aHvi1, load_B_fragT(sWl0T, DD, 32, n0, lane), l0c);
        r0c = WMMA_F16(aHvj0, load_B_fragT(sWr0T, DD, 0,  n0, lane), r0c);
        r0c = WMMA_F16(aHvj1, load_B_fragT(sWr0T, DD, 32, n0, lane), r0c);
        l1c = WMMA_F16(aHvi0, load_B_fragT(sWl1T, 2 * DD, 0,  n0, lane), l1c);
        l1c = WMMA_F16(aHvi1, load_B_fragT(sWl1T, 2 * DD, 32, n0, lane), l1c);
        l1c = WMMA_F16(aQe0,  load_B_fragT(sWl1T, 2 * DD, 64, n0, lane), l1c);
        l1c = WMMA_F16(aQe1,  load_B_fragT(sWl1T, 2 * DD, 96, n0, lane), l1c);
        r1c = WMMA_F16(aHvj0, load_B_fragT(sWr1T, 2 * DD, 0,  n0, lane), r1c);
        r1c = WMMA_F16(aHvj1, load_B_fragT(sWr1T, 2 * DD, 32, n0, lane), r1c);
        r1c = WMMA_F16(aQe0,  load_B_fragT(sWr1T, 2 * DD, 64, n0, lane), r1c);
        r1c = WMMA_F16(aQe1,  load_B_fragT(sWr1T, 2 * DD, 96, n0, lane), r1c);

        const int n = n0 + nloc;
        const float bl0n = sBia[n], br0n = sBia[DD + n];
        const float bl1n = sBia[2 * DD + n], br1n = sBia[3 * DD + n];
#pragma unroll
        for (int r = 0; r < 8; ++r) {
            const float L0 = fast_tanh(l0c[r] + bl0n), R0 = fast_tanh(r0c[r] + br0n);
            const float L1 = fast_tanh(l1c[r] + bl1n), R1 = fast_tanh(r1c[r] + br1n);
            partial[r] += L0 * R0 + L1 * R1;
        }
    }

    // Row-sum reduce over the 16 lanes of each half-wave (N dimension).
#pragma unroll
    for (int r = 0; r < 8; ++r) {
        float s = partial[r];
#pragma unroll
        for (int off = 8; off >= 1; off >>= 1) s += __shfl_xor(s, off, 16);
        if (nloc == 0) {
            const int e = ebase + r + 8 * g;
            if (e < nE) logits[e] = s;
        }
    }
}

// ---------------------------------------------------------------------------
// Segment softmax + scatter-add flow
// ---------------------------------------------------------------------------
__device__ __forceinline__ unsigned f2key(float f) {
    unsigned u = __float_as_uint(f);
    return (u & 0x80000000u) ? ~u : (u | 0x80000000u);
}
__device__ __forceinline__ float key2f(unsigned k) {
    unsigned u = (k & 0x80000000u) ? (k & 0x7fffffffu) : ~k;
    return __uint_as_float(u);
}

__global__ void seg_max_kernel(const float* __restrict__ logits, const int* __restrict__ edges,
                               unsigned* __restrict__ keys, int nE) {
    int e = blockIdx.x * blockDim.x + threadIdx.x;
    if (e >= nE) return;
    const int seg = edges[(long)e * 8 + 4];
    atomicMax(keys + seg, f2key(logits[e]));
}

__global__ void seg_exp_kernel(const float* __restrict__ logits, const int* __restrict__ edges,
                               const unsigned* __restrict__ keys, float* __restrict__ evals,
                               float* __restrict__ ssum, int nE) {
    int e = blockIdx.x * blockDim.x + threadIdx.x;
    if (e >= nE) return;
    const int seg = edges[(long)e * 8 + 4];
    const float ev = __expf(logits[e] - key2f(keys[seg]));
    evals[e] = ev;
    atomicAdd(ssum + seg, ev);
}

__global__ void flow_kernel(const float* __restrict__ evals, const float* __restrict__ ssum,
                            const int* __restrict__ edges, const float* __restrict__ edges_y,
                            const float* __restrict__ node_att, float* __restrict__ out, int nE) {
    int e = blockIdx.x * blockDim.x + threadIdx.x;
    if (e >= nE) return;
    const int* er = edges + (long)e * 8;
    const int eg = er[0], vi = er[1], vj = er[2], seg = er[4];
    const float trans = evals[e] * __builtin_amdgcn_rcpf(ssum[seg]) * edges_y[e];
    const float flow = node_att[(long)eg * NN + vi] * trans;
    atomicAdd(out + (long)eg * NN + vj, flow);
}

// ---------------------------------------------------------------------------
// Launcher
// ---------------------------------------------------------------------------
extern "C" void kernel_launch(void* const* d_in, const int* in_sizes, int n_in,
                              void* d_out, int out_size, void* d_ws, size_t ws_size,
                              hipStream_t stream) {
    const float* node_att = (const float*)d_in[0];
    const float* edges_y  = (const float*)d_in[1];
    const float* hid_un   = (const float*)d_in[2];   // (1, N, 256)
    const float* hid_con  = (const float*)d_in[3];   // (M_MEM, 64)
    const float* qh       = (const float*)d_in[4];
    const float* qr       = (const float*)d_in[5];
    const float* W_h      = (const float*)d_in[6];
    const float* b_h      = (const float*)d_in[7];
    const float* W_hlg    = (const float*)d_in[8];
    const float* b_hlg    = (const float*)d_in[9];
    const float* W_rlg    = (const float*)d_in[10];
    const float* b_rlg    = (const float*)d_in[11];
    const float* Wl0      = (const float*)d_in[12];
    const float* bl0      = (const float*)d_in[13];
    const float* Wr0      = (const float*)d_in[14];
    const float* br0      = (const float*)d_in[15];
    const float* Wl1      = (const float*)d_in[16];
    const float* bl1      = (const float*)d_in[17];
    const float* Wr1      = (const float*)d_in[18];
    const float* br1      = (const float*)d_in[19];
    const int*   edges    = (const int*)d_in[20];    // (E, 8)
    const int*   new_idx  = (const int*)d_in[21];    // (M_MEM, 1)
    float* out = (float*)d_out;

    // Workspace carve (floats)
    float* ws      = (float*)d_ws;
    float* hc_full = ws;                                   // MALL*64
    float* hu      = hc_full + (long)MALL * DD;            // NN*64
    float* qv      = hu + (long)NN * DD;                   // 32*64
    float* logits  = qv + BB * DD;                         // E
    float* evals   = logits + EE;                          // E
    unsigned* keys = (unsigned*)(evals + EE);              // NSEG
    float* ssum    = (float*)(keys + NSEG);                // NSEG

    // Zero accumulators (every call: graph replay must be deterministic)
    {
        int n;
        n = MALL * DD;  zero_kernel<<<(n + 255) / 256, 256, 0, stream>>>(hc_full, n);
        n = NSEG;       zero_kernel<<<(n + 255) / 256, 256, 0, stream>>>((float*)keys, n);
        n = NSEG;       zero_kernel<<<(n + 255) / 256, 256, 0, stream>>>(ssum, n);
        n = BB * NN;    zero_kernel<<<(n + 255) / 256, 256, 0, stream>>>(out, n);
    }

    // Query projections (tiny)
    qv_kernel<<<BB, DD, 0, stream>>>(qh, qr, W_hlg, b_hlg, W_rlg, b_rlg, qv);

    // hc scatter: tanh(hidden_con @ W_h + b_h) -> hc_full[new_idx]
    gemm_tanh_kernel<DD, 4><<<(MMEM + 63) / 64, 128, 0, stream>>>(
        hid_con, W_h, b_h, new_idx, hc_full, MMEM);

    // hu: tanh(hidden_uncon @ W_hlg + b_hlg)
    gemm_tanh_kernel<DLG, 2><<<(NN + 31) / 32, 64, 0, stream>>>(
        hid_un, W_hlg, b_hlg, nullptr, hu, NN);

    // Fused per-edge WMMA scoring
    {
        const int edges_per_block = EWAVES * 16;
        const int blocks = (EE + edges_per_block - 1) / edges_per_block;
        const size_t smem = (size_t)(2 * DD * DD + 2 * 2 * DD * DD) * sizeof(_Float16)
                          + (size_t)(4 * DD) * sizeof(float)
                          + (size_t)(EWAVES * 3 * 16 * DD) * sizeof(_Float16);
        edge_logits_kernel<<<blocks, EWAVES * 32, smem, stream>>>(
            edges, hc_full, hu, qv, Wl0, bl0, Wr0, br0, Wl1, bl1, Wr1, br1, logits, EE);
    }

    // Segment softmax + flow scatter
    const int tb = 256, gb = (EE + tb - 1) / tb;
    seg_max_kernel<<<gb, tb, 0, stream>>>(logits, edges, keys, EE);
    seg_exp_kernel<<<gb, tb, 0, stream>>>(logits, edges, keys, evals, ssum, EE);
    flow_kernel<<<gb, tb, 0, stream>>>(evals, ssum, edges, edges_y, node_att, out, EE);
}
